// FastLocalUnitaryGCNConv_53807350284428
// MI455X (gfx1250) — matrix operand order repaired and checked by reading, not compiled, not run on hardware
//
#include <hip/hip_runtime.h>

// Problem constants (match reference)
#define N_NODES 20000
#define F_IN    512
#define F_OUT   512
#define F2      1024           // [real | imag] concatenated GEMM width
#define SS      6
#define HT      0.025f         // T * HSCALE
#define MAX_TERMS 4
#define BN_EPS  1e-5f

typedef unsigned short ushort_t;
typedef __attribute__((ext_vector_type(16))) __bf16 v16bf;
typedef __attribute__((ext_vector_type(8)))  float  v8f;

union FragBF { v16bf v; uint4 q[2]; };

__device__ __forceinline__ ushort_t f2bf(float f) {
    unsigned u = __float_as_uint(f);
    unsigned r = (u + 0x7FFFu + ((u >> 16) & 1u)) >> 16;   // round-nearest-even
    return (ushort_t)r;
}
__device__ __forceinline__ float bf2f(ushort_t h) {
    return __uint_as_float(((unsigned)h) << 16);
}

// ---------------------------------------------------------------------------
// Kernel 0: x -> bf16; W_re/W_im -> bf16 *transposed* Wt[F2][F_IN]; zero stats
// Transposing W once lets the GEMM stage BOTH tiles as straight b128 copies.
// ---------------------------------------------------------------------------
__global__ void prep_kernel(const float* __restrict__ x,
                            const float* __restrict__ Wre,
                            const float* __restrict__ Wim,
                            ushort_t* __restrict__ xb,
                            ushort_t* __restrict__ Wt,   // [F2][F_IN]
                            float* __restrict__ stats)
{
    int idx    = blockIdx.x * blockDim.x + threadIdx.x;
    int stride = gridDim.x * blockDim.x;
    for (int i = idx; i < N_NODES * F_IN; i += stride)
        xb[i] = f2bf(x[i]);
    for (int i = idx; i < F_IN * F_OUT; i += stride) {
        int k = i / F_OUT, c = i % F_OUT;
        Wt[(size_t)c * F_IN + k]           = f2bf(Wre[i]);
        Wt[(size_t)(F_OUT + c) * F_IN + k] = f2bf(Wim[i]);
    }
    for (int i = idx; i < 4 * F_OUT; i += stride) stats[i] = 0.f;
}

// ---------------------------------------------------------------------------
// Kernel 1: yb[N,1024](bf16) = xb[N,512] @ W + bias   (W given transposed)
// Block 256 thr (8 waves as 4Mx2N), tile 128x128, K-chunk 32, double-buffered
// LDS, 8 WMMAs per wave per K-chunk (2 A-frags x 4 B-frags).
// ---------------------------------------------------------------------------
#define BM 128
#define BN 128
#define BK 32
#define KITERS (F_IN / BK)     // 16

__global__ __launch_bounds__(256)
void gemm_bf16_kernel(const ushort_t* __restrict__ A,   // [N_NODES][F_IN]
                      const ushort_t* __restrict__ Wt,  // [F2][F_IN]
                      const float* __restrict__ bre,
                      const float* __restrict__ bim,
                      ushort_t* __restrict__ C)         // [N_NODES][F2]
{
    // double-buffered tiles; both K-major rows of 32 halves (64 B)
    __shared__ __align__(16) ushort_t As[2][BM][BK];    // 16 KB
    __shared__ __align__(16) ushort_t Bs[2][BN][BK];    // 16 KB

    const int tid  = threadIdx.x;
    const int lane = tid & 31;
    const int wv   = tid >> 5;          // 0..7
    const int wm   = wv & 3;            // wave M index: 32 rows each
    const int wn   = wv >> 2;           // wave N index: 64 cols each
    const int m0   = blockIdx.x * BM;
    const int n0   = blockIdx.y * BN;
    const int r    = lane & 15;
    const int hi   = lane >> 4;

    v8f acc[2][4] = {};

    // per-thread staging indices: 512 uint4 per tile, 2 per thread
    const int i0   = tid * 2;
    const int row0 = i0 >> 2;           // 4 uint4 per 32-half row
    const int k0   = (i0 & 3) * 8;
    const int row1 = (i0 + 1) >> 2;
    const int k1   = ((i0 + 1) & 3) * 8;

    auto ldA = [&](int kc, int rrow, int koff) -> uint4 {
        int grow = m0 + rrow; if (grow >= N_NODES) grow = N_NODES - 1;
        return *(const uint4*)(A + (size_t)grow * F_IN + kc + koff);
    };
    auto ldB = [&](int kc, int rrow, int koff) -> uint4 {
        return *(const uint4*)(Wt + (size_t)(n0 + rrow) * F_IN + kc + koff);
    };

    // prologue: stage K-chunk 0 into buffer 0
    {
        uint4 a0 = ldA(0, row0, k0), a1 = ldA(0, row1, k1);
        uint4 b0 = ldB(0, row0, k0), b1 = ldB(0, row1, k1);
        ((uint4*)As[0])[i0]     = a0; ((uint4*)As[0])[i0 + 1] = a1;
        ((uint4*)Bs[0])[i0]     = b0; ((uint4*)Bs[0])[i0 + 1] = b1;
    }
    __syncthreads();

    for (int it = 0; it < KITERS; ++it) {
        const int cur = it & 1;
        const int nxt = cur ^ 1;

        // issue next tile's global loads first (overlap with WMMA below)
        uint4 a0, a1, b0, b1;
        const bool have_next = (it + 1) < KITERS;
        if (have_next) {
            int kc = (it + 1) * BK;
            a0 = ldA(kc, row0, k0); a1 = ldA(kc, row1, k1);
            b0 = ldB(kc, row0, k0); b1 = ldB(kc, row1, k1);
        }

        // fragments + 8 WMMAs from current buffer
        FragBF fa[2];
        #pragma unroll
        for (int mi = 0; mi < 2; ++mi) {
            int arow = wm * 32 + mi * 16 + r;
            fa[mi].q[0] = *(const uint4*)&As[cur][arow][hi * 8];
            fa[mi].q[1] = *(const uint4*)&As[cur][arow][hi * 8 + 16];
        }
        #pragma unroll
        for (int t = 0; t < 4; ++t) {
            FragBF fb;
            int bcol = wn * 64 + t * 16 + r;
            fb.q[0] = *(const uint4*)&Bs[cur][bcol][hi * 8];
            fb.q[1] = *(const uint4*)&Bs[cur][bcol][hi * 8 + 16];
            acc[0][t] = __builtin_amdgcn_wmma_f32_16x16x32_bf16(
                false, fa[0].v, false, fb.v, (short)0, acc[0][t], false, false);
            acc[1][t] = __builtin_amdgcn_wmma_f32_16x16x32_bf16(
                false, fa[1].v, false, fb.v, (short)0, acc[1][t], false, false);
        }

        // store next tile (waits on the global loads), then one barrier
        if (have_next) {
            ((uint4*)As[nxt])[i0]     = a0; ((uint4*)As[nxt])[i0 + 1] = a1;
            ((uint4*)Bs[nxt])[i0]     = b0; ((uint4*)Bs[nxt])[i0 + 1] = b1;
        }
        __syncthreads();
    }

    // Epilogue: C/D layout -> M = j + (lane>=16 ? 8 : 0), N = lane & 15
    #pragma unroll
    for (int mi = 0; mi < 2; ++mi) {
        #pragma unroll
        for (int t = 0; t < 4; ++t) {
            int col = n0 + wn * 64 + t * 16 + r;
            float bias = (col < F_OUT) ? bre[col] : bim[col - F_OUT];
            #pragma unroll
            for (int j = 0; j < 8; ++j) {
                int grow = m0 + wm * 32 + mi * 16 + j + hi * 8;
                if (grow < N_NODES)
                    C[(size_t)grow * F2 + col] = f2bf(acc[mi][t][j] + bias);
            }
        }
    }
}

// ---------------------------------------------------------------------------
// Kernel 2: per-node 6x6 Laplacian exp-series row 0, complex gather-combine,
// ReLU, write er|ei into d_out, accumulate BN partial sums.
// Block = 256 threads, 64 nodes per block; threads 0..63 compute res -> LDS.
// ---------------------------------------------------------------------------
#define CHUNK 64

__global__ __launch_bounds__(256)
void evolve_kernel(const ushort_t* __restrict__ yb,      // [N][F2] bf16
                   const int* __restrict__ sub_nodes,    // [N][6]
                   const int* __restrict__ sub_adj,      // [N][36]
                   float* __restrict__ out,              // [N][F2] er|ei pre-BN
                   float* __restrict__ stats)            // sum_r, ss_r, sum_i, ss_i
{
    __shared__ float rr[CHUNK][SS];
    __shared__ float ri[CHUNK][SS];
    __shared__ int   gs[CHUNK][SS];
    const int tid  = threadIdx.x;
    const int base = blockIdx.x * CHUNK;

    if (tid < CHUNK) {
        int n = base + tid;
        if (n < N_NODES) {
            const int* adj = sub_adj + (size_t)n * (SS * SS);
            const int* g   = sub_nodes + (size_t)n * SS;
            #pragma unroll
            for (int s = 0; s < SS; ++s) gs[tid][s] = g[s];

            float L[SS][SS];
            #pragma unroll
            for (int s = 0; s < SS; ++s) {
                int deg = 0;
                #pragma unroll
                for (int t = 0; t < SS; ++t) deg += adj[s * SS + t];
                #pragma unroll
                for (int t = 0; t < SS; ++t)
                    L[s][t] = (float)((s == t) ? deg : 0) - (float)adj[s * SS + t];
            }
            // term_k = term_{k-1} @ (c*L) / k  with c = -i*HT
            float tr[SS], ti[SS], resr[SS], resi[SS];
            #pragma unroll
            for (int s = 0; s < SS; ++s) {
                tr[s] = (s == 0) ? 1.f : 0.f; ti[s] = 0.f;
                resr[s] = tr[s]; resi[s] = 0.f;
            }
            #pragma unroll
            for (int k = 1; k <= MAX_TERMS; ++k) {
                float ur[SS], ui[SS];
                #pragma unroll
                for (int t = 0; t < SS; ++t) {
                    float ar = 0.f, ai = 0.f;
                    #pragma unroll
                    for (int s = 0; s < SS; ++s) {
                        ar += tr[s] * L[s][t];
                        ai += ti[s] * L[s][t];
                    }
                    ur[t] = ar; ui[t] = ai;
                }
                float fk = HT / (float)k;
                #pragma unroll
                for (int t = 0; t < SS; ++t) {
                    tr[t] =  fk * ui[t];     // (u)*(-i*h/k): re =  h/k * ui
                    ti[t] = -fk * ur[t];     //              im = -h/k * ur
                    resr[t] += tr[t]; resi[t] += ti[t];
                }
            }
            #pragma unroll
            for (int s = 0; s < SS; ++s) { rr[tid][s] = resr[s]; ri[tid][s] = resi[s]; }
        }
    }
    __syncthreads();

    const int f = tid * 2;                               // features f, f+1
    float sr0=0, sr1=0, qr0=0, qr1=0, si0=0, si1=0, qi0=0, qi1=0;
    int nmax = N_NODES - base; if (nmax > CHUNK) nmax = CHUNK;

    for (int j = 0; j < nmax; ++j) {
        int n = base + j;
        float er0=0, er1=0, ei0=0, ei1=0;
        #pragma unroll
        for (int s = 0; s < SS; ++s) {
            size_t rowb = (size_t)gs[j][s] * F2;
            unsigned pr = *(const unsigned*)(yb + rowb + f);
            unsigned pi = *(const unsigned*)(yb + rowb + F_OUT + f);
            float yr0 = bf2f((ushort_t)(pr & 0xFFFFu)), yr1 = bf2f((ushort_t)(pr >> 16));
            float yi0 = bf2f((ushort_t)(pi & 0xFFFFu)), yi1 = bf2f((ushort_t)(pi >> 16));
            float cr = rr[j][s], ci = ri[j][s];
            er0 += cr * yr0 - ci * yi0;  er1 += cr * yr1 - ci * yi1;
            ei0 += cr * yi0 + ci * yr0;  ei1 += cr * yi1 + ci * yr1;
        }
        er0 = fmaxf(er0, 0.f); er1 = fmaxf(er1, 0.f);
        ei0 = fmaxf(ei0, 0.f); ei1 = fmaxf(ei1, 0.f);
        float* orow = out + (size_t)n * F2;
        orow[f] = er0; orow[f + 1] = er1;
        orow[F_OUT + f] = ei0; orow[F_OUT + f + 1] = ei1;
        sr0 += er0; sr1 += er1; qr0 += er0 * er0; qr1 += er1 * er1;
        si0 += ei0; si1 += ei1; qi0 += ei0 * ei0; qi1 += ei1 * ei1;
    }

    float* sum_r = stats;
    float* ss_r  = stats + F_OUT;
    float* sum_i = stats + 2 * F_OUT;
    float* ss_i  = stats + 3 * F_OUT;
    atomicAdd(&sum_r[f],     sr0); atomicAdd(&sum_r[f + 1], sr1);
    atomicAdd(&ss_r[f],      qr0); atomicAdd(&ss_r[f + 1],  qr1);
    atomicAdd(&sum_i[f],     si0); atomicAdd(&sum_i[f + 1], si1);
    atomicAdd(&ss_i[f],      qi0); atomicAdd(&ss_i[f + 1],  qi1);
}

// ---------------------------------------------------------------------------
// Kernel 3: turn sums into per-feature (scale, shift) for real/imag BN
// ---------------------------------------------------------------------------
__global__ void bn_finalize_kernel(const float* __restrict__ stats,
                                   const float* __restrict__ gr, const float* __restrict__ br,
                                   const float* __restrict__ gi, const float* __restrict__ bi,
                                   float* __restrict__ coef)    // scale_r, shift_r, scale_i, shift_i
{
    int fidx = blockIdx.x * blockDim.x + threadIdx.x;
    if (fidx >= F_OUT) return;
    const float invN = 1.0f / (float)N_NODES;
    float mr  = stats[fidx] * invN;
    float vr  = stats[F_OUT + fidx] * invN - mr * mr;
    float scr = gr[fidx] * rsqrtf(vr + BN_EPS);
    coef[fidx]          = scr;
    coef[F_OUT + fidx]  = br[fidx] - mr * scr;
    float mi  = stats[2 * F_OUT + fidx] * invN;
    float vi  = stats[3 * F_OUT + fidx] * invN - mi * mi;
    float sci = gi[fidx] * rsqrtf(vi + BN_EPS);
    coef[2 * F_OUT + fidx] = sci;
    coef[3 * F_OUT + fidx] = bi[fidx] - mi * sci;
}

// ---------------------------------------------------------------------------
// Kernel 4: in-place BN apply + residual (real half only gets +x)
// ---------------------------------------------------------------------------
__global__ void bn_apply_kernel(const float* __restrict__ x,
                                const float* __restrict__ coef,
                                float* __restrict__ out)
{
    size_t idx    = (size_t)blockIdx.x * blockDim.x + threadIdx.x;
    size_t stride = (size_t)gridDim.x * blockDim.x;
    const size_t total = (size_t)N_NODES * F2;
    for (size_t i = idx; i < total; i += stride) {
        int c = (int)(i % F2);
        size_t n = i / F2;
        float v = out[i];
        if (c < F_OUT) {
            out[i] = v * coef[c] + coef[F_OUT + c] + x[n * F_IN + c];
        } else {
            int ci = c - F_OUT;
            out[i] = v * coef[2 * F_OUT + ci] + coef[3 * F_OUT + ci];
        }
    }
}

// ---------------------------------------------------------------------------
extern "C" void kernel_launch(void* const* d_in, const int* in_sizes, int n_in,
                              void* d_out, int out_size, void* d_ws, size_t ws_size,
                              hipStream_t stream)
{
    const float* x         = (const float*)d_in[0];
    // d_in[1] = edge_index (unused; subgraphs precomputed)
    const int*   sub_nodes = (const int*)d_in[2];
    const int*   sub_adj   = (const int*)d_in[3];
    const float* Wre       = (const float*)d_in[4];
    const float* Wim       = (const float*)d_in[5];
    const float* bre       = (const float*)d_in[6];
    const float* bim       = (const float*)d_in[7];
    const float* gr        = (const float*)d_in[8];
    const float* br        = (const float*)d_in[9];
    const float* gi        = (const float*)d_in[10];
    const float* bi        = (const float*)d_in[11];

    char* p = (char*)d_ws;
    auto take = [&](size_t bytes) -> char* {
        char* q = p;
        p += (bytes + 255) & ~(size_t)255;
        return q;
    };
    ushort_t* xb    = (ushort_t*)take((size_t)N_NODES * F_IN * 2);   // ~20.5 MB
    ushort_t* Wt    = (ushort_t*)take((size_t)F2 * F_IN * 2);        //  ~1 MB
    ushort_t* yb    = (ushort_t*)take((size_t)N_NODES * F2 * 2);     // ~41 MB (L2-resident)
    float*    stats = (float*)take((size_t)4 * F_OUT * 4);
    float*    coef  = (float*)take((size_t)4 * F_OUT * 4);

    float* out = (float*)d_out;

    prep_kernel<<<dim3(8192), dim3(256), 0, stream>>>(x, Wre, Wim, xb, Wt, stats);

    dim3 ggrid((N_NODES + BM - 1) / BM, F2 / BN);
    gemm_bf16_kernel<<<ggrid, dim3(256), 0, stream>>>(xb, Wt, bre, bim, yb);

    evolve_kernel<<<dim3((N_NODES + CHUNK - 1) / CHUNK), dim3(256), 0, stream>>>(
        yb, sub_nodes, sub_adj, out, stats);

    bn_finalize_kernel<<<dim3(1), dim3(512), 0, stream>>>(stats, gr, br, gi, bi, coef);

    bn_apply_kernel<<<dim3(8192), dim3(256), 0, stream>>>(x, coef, out);
}
